// TritonLinearW8A8Impl_78872779424013
// MI455X (gfx1250) — 8-bit matrix operand order repaired and checked by reading, compiled, not run on hardware
//
#include <hip/hip_runtime.h>

typedef __attribute__((ext_vector_type(8)))  int      v8i;
typedef __attribute__((ext_vector_type(4)))  int      v4i;
typedef __attribute__((ext_vector_type(2)))  int      v2i;
typedef __attribute__((ext_vector_type(8)))  _Float16 v8h;
typedef _Float16 half_t;

#define MDIM 4096
#define KDIM 4096
#define NDIM 11008
#define QEPS 1e-7f

#define KSTEP    128
#define ROWPAD   144
#define A_BYTES  (128 * ROWPAD)
#define B_BYTES  (256 * ROWPAD)
#define BUF_BYTES (A_BYTES + B_BYTES)
#define NSTAGE   (KDIM / KSTEP)
#define ROWSKIP  (32 * KDIM)

template <int IMM>
__device__ __forceinline__ void async_b128(unsigned lds_off, const void* gaddr)
{
    asm volatile("global_load_async_to_lds_b128 %0, %1, off offset:%c2"
                 :: "v"(lds_off), "v"(gaddr), "n"(IMM) : "memory");
}

__device__ __forceinline__ void wait_asynccnt0()
{
#if __has_builtin(__builtin_amdgcn_s_wait_asynccnt)
    __builtin_amdgcn_s_wait_asynccnt(0);
#else
    asm volatile("s_wait_asynccnt 0" ::: "memory");
#endif
}

__global__ __launch_bounds__(256) void quant_rows(const half_t* __restrict__ x,
                                                  signed char* __restrict__ q,
                                                  float* __restrict__ rscale)
{
    __shared__ float red[256];
    const int row  = blockIdx.x;
    const int t    = threadIdx.x;
    const half_t* xr = x + (size_t)row * KDIM;
    const int base = t * 16;

    v8h h0 = *(const v8h*)(xr + base);
    v8h h1 = *(const v8h*)(xr + base + 8);

    float f[16];
    float amax = 0.0f;
#pragma unroll
    for (int i = 0; i < 8; ++i) { f[i] = (float)h0[i]; f[i + 8] = (float)h1[i]; }
#pragma unroll
    for (int i = 0; i < 16; ++i) amax = fmaxf(amax, fabsf(f[i]));

    red[t] = amax;
    __syncthreads();
#pragma unroll
    for (int s = 128; s > 0; s >>= 1) {
        if (t < s) red[t] = fmaxf(red[t], red[t + s]);
        __syncthreads();
    }
    const float scale = fmaxf(red[0], QEPS) * (1.0f / 127.0f);
    const float inv   = 1.0f / scale;
    if (t == 0) rscale[row] = scale;

    int qi[16];
#pragma unroll
    for (int i = 0; i < 16; ++i) {
        float v = rintf(f[i] * inv);
        v = fminf(fmaxf(v, -128.0f), 127.0f);
        qi[i] = (int)v;
    }
    v4i packed;
#pragma unroll
    for (int w = 0; w < 4; ++w)
        packed[w] = (qi[4 * w] & 255) | ((qi[4 * w + 1] & 255) << 8) |
                    ((qi[4 * w + 2] & 255) << 16) | ((qi[4 * w + 3] & 255) << 24);
    *(v4i*)(q + (size_t)row * KDIM + base) = packed;
}

__global__ __launch_bounds__(256, 1) void wmma_gemm(
    const signed char* __restrict__ q,
    const signed char* __restrict__ w,
    const float* __restrict__ rscale,
    const float* __restrict__ wscale,
    const float* __restrict__ bias,
    half_t* __restrict__ out)
{
    __shared__ __attribute__((aligned(16))) char smem[2 * BUF_BYTES];

    const int t     = threadIdx.x;
    const int lane  = t & 31;
    const int wid   = t >> 5;
    const int waveM = wid >> 2;
    const int waveN = wid & 3;
    const int bm0 = blockIdx.y * 128;
    const int bn0 = blockIdx.x * 256;
    const int bm  = bm0 + waveM * 64;
    const int bn  = bn0 + waveN * 64;

    const int lo = lane & 15;
    const int hi = lane >> 4;

    const unsigned smb = (unsigned)(unsigned long long)(void*)smem;

    const int crow = t >> 3;
    const int ckp  = (t & 7) << 4;
    const signed char* aP = q + (size_t)(bm0 + crow) * KDIM + ckp;
    const signed char* bP = w + (size_t)(bn0 + crow) * KDIM + ckp;
    const unsigned aldsrel = smb + (unsigned)(crow * ROWPAD + ckp);

    v8i acc[4][4];
#pragma unroll
    for (int r = 0; r < 4; ++r)
#pragma unroll
        for (int c = 0; c < 4; ++c)
#pragma unroll
            for (int e = 0; e < 8; ++e) acc[r][c][e] = 0;

#define ISSUE_STAGE(BUFBASE, APTR, BPTR)                                        \
    do {                                                                        \
        const unsigned al = (BUFBASE) + aldsrel;                                \
        const unsigned bl = (BUFBASE) + aldsrel + A_BYTES;                      \
        async_b128<0 * ROWSKIP>(al + 0 * 32 * ROWPAD, (APTR));                  \
        async_b128<1 * ROWSKIP>(al + 1 * 32 * ROWPAD, (APTR));                  \
        async_b128<2 * ROWSKIP>(al + 2 * 32 * ROWPAD, (APTR));                  \
        async_b128<3 * ROWSKIP>(al + 3 * 32 * ROWPAD, (APTR));                  \
        async_b128<0 * ROWSKIP>(bl + 0 * 32 * ROWPAD, (BPTR));                  \
        async_b128<1 * ROWSKIP>(bl + 1 * 32 * ROWPAD, (BPTR));                  \
        async_b128<2 * ROWSKIP>(bl + 2 * 32 * ROWPAD, (BPTR));                  \
        async_b128<3 * ROWSKIP>(bl + 3 * 32 * ROWPAD, (BPTR));                  \
        async_b128<4 * ROWSKIP>(bl + 4 * 32 * ROWPAD, (BPTR));                  \
        async_b128<5 * ROWSKIP>(bl + 5 * 32 * ROWPAD, (BPTR));                  \
        async_b128<6 * ROWSKIP>(bl + 6 * 32 * ROWPAD, (BPTR));                  \
        async_b128<7 * ROWSKIP>(bl + 7 * 32 * ROWPAD, (BPTR));                  \
    } while (0)

    ISSUE_STAGE(0u, aP, bP);

    const unsigned afrag = (unsigned)((waveM * 64 + lo) * ROWPAD + hi * 16);
    const unsigned bfrag = (unsigned)(A_BYTES + (waveN * 64 + lo) * ROWPAD + hi * 16);

    for (int s = 0; s < NSTAGE; ++s) {
        wait_asynccnt0();
        __syncthreads();

        if (s + 1 < NSTAGE) {
            const unsigned nb = (unsigned)(((s + 1) & 1) * BUF_BYTES);
            const int kk = (s + 1) * KSTEP;
            ISSUE_STAGE(nb, aP + kk, bP + kk);
        }

        const unsigned bufb = (unsigned)((s & 1) * BUF_BYTES);
#pragma unroll
        for (int kk2 = 0; kk2 < 2; ++kk2) {
            const char* ab = (const char*)smem + (bufb + afrag + kk2 * 64);
            const char* bb = (const char*)smem + (bufb + bfrag + kk2 * 64);

            v8i b[4];
#pragma unroll
            for (int c = 0; c < 4; ++c) {
                const char* p = bb + c * 16 * ROWPAD;
                v4i e0 = *(const v4i*)(p);
                v4i e1 = *(const v4i*)(p + 32);
                b[c][0] = e0[0]; b[c][1] = e0[1]; b[c][2] = e0[2]; b[c][3] = e0[3];
                b[c][4] = e1[0]; b[c][5] = e1[1]; b[c][6] = e1[2]; b[c][7] = e1[3];
            }

            v8i acur, anext;
            {
                const char* p = ab;
                v2i d0 = *(const v2i*)(p);
                v2i d1 = *(const v2i*)(p + 8);
                v2i d2 = *(const v2i*)(p + 32);
                v2i d3 = *(const v2i*)(p + 40);
                acur[0] = d0[0]; acur[1] = d0[1]; acur[2] = d1[0]; acur[3] = d1[1];
                acur[4] = d2[0]; acur[5] = d2[1]; acur[6] = d3[0]; acur[7] = d3[1];
            }
#pragma unroll
            for (int r = 0; r < 4; ++r) {
                if (r < 3) {
                    const char* p = ab + (r + 1) * 16 * ROWPAD;
                    v2i d0 = *(const v2i*)(p);
                    v2i d1 = *(const v2i*)(p + 8);
                    v2i d2 = *(const v2i*)(p + 32);
                    v2i d3 = *(const v2i*)(p + 40);
                    anext[0] = d0[0]; anext[1] = d0[1];
                    anext[2] = d1[0]; anext[3] = d1[1];
                    anext[4] = d2[0]; anext[5] = d2[1];
                    anext[6] = d3[0]; anext[7] = d3[1];
                }
#pragma unroll
                for (int c = 0; c < 4; ++c)
                    acc[r][c] = __builtin_amdgcn_wmma_i32_16x16x64_iu8(
                        true, acur, true, b[c], acc[r][c], false, false);
                acur = anext;
            }
        }
    }

#pragma unroll
    for (int c = 0; c < 4; ++c) {
        const int col = bn + c * 16 + lo;
        const float ws = wscale[col];
        const float bs = bias[col];
#pragma unroll
        for (int r = 0; r < 4; ++r) {
#pragma unroll
            for (int e = 0; e < 8; ++e) {
                const int row = bm + r * 16 + (hi << 3) + e;
                const float v = (float)acc[r][c][e] * rscale[row] * ws + bs;
                out[(size_t)row * NDIM + col] = (half_t)v;
            }
        }
    }
}

extern "C" void kernel_launch(void* const* d_in, const int* in_sizes, int n_in,
                              void* d_out, int out_size, void* d_ws, size_t ws_size,
                              hipStream_t stream)
{
    const half_t*      x      = (const half_t*)d_in[0];
    const signed char* weight = (const signed char*)d_in[1];
    const float*       wscale = (const float*)d_in[2];
    const float*       bias   = (const float*)d_in[3];
    half_t*            out    = (half_t*)d_out;

    signed char* q      = (signed char*)d_ws;
    float*       rscale = (float*)((char*)d_ws + (size_t)MDIM * KDIM);

    quant_rows<<<MDIM, 256, 0, stream>>>(x, q, rscale);

    dim3 grid(NDIM / 256, MDIM / 128);
    wmma_gemm<<<grid, 256, 0, stream>>>(q, weight, rscale, wscale, bias, out);
}
